// GraniteMoeMoE_25744033972492
// MI455X (gfx1250) — matrix operand order repaired and checked by reading, not compiled, not run on hardware
//
#include <hip/hip_runtime.h>

#define NUM_EXPERTS 32
#define TOP_K 8
#define HIDDEN 1536
#define INTER 512
#define M_TILE 64
#define MSUB (M_TILE / 16)          // 4 m-subtiles
#define XS_STRIDE 1544   // halves; 3088B row stride -> 772 dwords, 772%64=4: conflict-free
#define ACT_STRIDE 520   // halves; 1040B row stride -> 260 dwords, 260%64=4: conflict-free

typedef __attribute__((ext_vector_type(16))) __bf16 v16bf;
typedef __attribute__((ext_vector_type(8)))  __bf16 v8bf;
typedef __attribute__((ext_vector_type(4)))  __bf16 v4bf;
typedef __attribute__((ext_vector_type(8)))  float  v8f;

// ---------------- fragment loaders ----------------

// A-fragment (16x32 bf16, M x K) given a per-subtile LDS base that already
// includes the lane's row offset (laneLo*stride) and the laneHi K-phase.
// Contiguous k order (matches B): lane L<16 holds row M=L, halves = K{0..15}; lane>=16: K{16..31}.
__device__ __forceinline__ v16bf load_a_frag_at(const __bf16* laneBase, int kk) {
    const __bf16* p = laneBase + kk * 32;
    v8bf lo = *(const v8bf*)(p);
    v8bf hi = *(const v8bf*)(p + 8);
    v16bf a;
#pragma unroll
    for (int i = 0; i < 8; ++i) { a[i] = lo[i]; a[i + 8] = hi[i]; }
    return a;
}

// B-fragment (32x16, K x N) where B[k][n] = W[n][k], W row-major with leading dim ld.
// Contiguous k order (matches A): lanes 0-15 hold N=lane, K=0..15; lanes 16-31 hold N=lane-16, K=16..31.
__device__ __forceinline__ v16bf load_b_frag(const float* wbase, int ld, int kk,
                                             int laneLo, int laneHi) {
    const float4* p4 = (const float4*)(wbase + (size_t)laneLo * ld + kk * 32 + laneHi * 16);
    v16bf b;
#pragma unroll
    for (int i = 0; i < 4; ++i) {
        float4 f = p4[i];
        b[i * 4 + 0] = (__bf16)f.x; b[i * 4 + 1] = (__bf16)f.y;
        b[i * 4 + 2] = (__bf16)f.z; b[i * 4 + 3] = (__bf16)f.w;
    }
    return b;
}

__device__ __forceinline__ v16bf load_b_frag(const __bf16* wbase, int ld, int kk,
                                             int laneLo, int laneHi) {
    const v8bf* p = (const v8bf*)(wbase + (size_t)laneLo * ld + kk * 32 + laneHi * 16);
    v8bf lo = p[0], hi = p[1];
    v16bf b;
#pragma unroll
    for (int i = 0; i < 8; ++i) { b[i] = lo[i]; b[i + 8] = hi[i]; }
    return b;
}

// ---------------- small kernels ----------------

__global__ void moe_zero_kernel(float* __restrict__ out, size_t n, int* __restrict__ counts) {
    size_t i = (size_t)blockIdx.x * blockDim.x + threadIdx.x;
    if (i < n) out[i] = 0.f;
    if (blockIdx.x == 0 && threadIdx.x < NUM_EXPERTS) counts[threadIdx.x] = 0;
}

// fp32 -> bf16 bulk convert, 8 elements/thread
__global__ void moe_cvt_bf16_kernel(__bf16* __restrict__ dst, const float* __restrict__ src, size_t n8) {
    size_t i = (size_t)blockIdx.x * blockDim.x + threadIdx.x;
    if (i >= n8) return;
    const float4* s = (const float4*)(src + i * 8);
    float4 f0 = s[0], f1 = s[1];
    v8bf b;
    b[0] = (__bf16)f0.x; b[1] = (__bf16)f0.y; b[2] = (__bf16)f0.z; b[3] = (__bf16)f0.w;
    b[4] = (__bf16)f1.x; b[5] = (__bf16)f1.y; b[6] = (__bf16)f1.z; b[7] = (__bf16)f1.w;
    *(v8bf*)(dst + i * 8) = b;
}

// One wave per token: logits, softmax, top-8 (min-index tiebreak), renormalize,
// atomic-append (token, weight) onto each selected expert's list.
__global__ __launch_bounds__(256) void moe_router_kernel(
    const float* __restrict__ x, const float* __restrict__ gate_w,
    int* __restrict__ counts, int* __restrict__ tok_id, float* __restrict__ tok_w, int T)
{
    const int wave = threadIdx.x >> 5;
    const int lane = threadIdx.x & 31;
    const int t = blockIdx.x * 8 + wave;
    if (t >= T) return;

    const float4* xr = (const float4*)(x + (size_t)t * HIDDEN);
    const float4* gr = (const float4*)(gate_w + (size_t)lane * HIDDEN);
    float acc = 0.f;
    for (int i = 0; i < HIDDEN / 4; ++i) {
        float4 a = xr[i], b = gr[i];
        acc += a.x * b.x + a.y * b.y + a.z * b.z + a.w * b.w;
    }
    float m = acc;
    for (int o = 16; o; o >>= 1) m = fmaxf(m, __shfl_xor(m, o));
    float p = __expf(acc - m);
    float s = p;
    for (int o = 16; o; o >>= 1) s += __shfl_xor(s, o);
    float prob = p / s;

    float v = prob, sumtop = 0.f;
    int sel = 0;
    for (int it = 0; it < TOP_K; ++it) {
        float bv = v; int bi = lane;
        for (int o = 16; o; o >>= 1) {
            float ov = __shfl_xor(bv, o);
            int   oi = __shfl_xor(bi, o);
            if (ov > bv || (ov == bv && oi < bi)) { bv = ov; bi = oi; }
        }
        sumtop += bv;
        if (lane == bi) { sel = 1; v = -1.f; }
    }
    if (sel) {
        float wgt = prob / sumtop;
        int pos = atomicAdd(&counts[lane], 1);
        tok_id[lane * T + pos] = t;
        tok_w [lane * T + pos] = wgt;
    }
}

// ---------------- fused expert FFN ----------------
// One (expert, 64-token tile) per 256-thread block (8 waves).
// XT: gathered activation source type (float or __bf16)
// WT: weight source type (float or __bf16)
template <typename XT, typename WT>
__global__ __launch_bounds__(256) void moe_expert_kernel(
    const XT* __restrict__ x, const WT* __restrict__ w13, const WT* __restrict__ w2,
    const int* __restrict__ counts, const int* __restrict__ tok_id, const float* __restrict__ tok_w,
    float* __restrict__ out, int T)
{
    extern __shared__ char smem[];
    __bf16* xs   = (__bf16*)smem;                                        // M_TILE x XS_STRIDE
    __bf16* acts = (__bf16*)(smem + M_TILE * XS_STRIDE * 2);             // M_TILE x ACT_STRIDE
    int*    lt   = (int*)  (smem + M_TILE * XS_STRIDE * 2 + M_TILE * ACT_STRIDE * 2);
    float*  lw   = (float*)(lt + M_TILE);

    const int e   = blockIdx.y;
    const int cnt = counts[e];
    const int m0  = blockIdx.x * M_TILE;
    if (m0 >= cnt) return;
    const int rows = min(M_TILE, cnt - m0);

    const int tid    = threadIdx.x;
    const int wave   = tid >> 5;
    const int lane   = tid & 31;
    const int laneLo = lane & 15;
    const int laneHi = lane >> 4;

    if (tid < M_TILE) {
        int t = -1; float wv = 0.f;
        if (tid < rows) { t = tok_id[e * T + m0 + tid]; wv = tok_w[e * T + m0 + tid]; }
        lt[tid] = t; lw[tid] = wv;
    }
    // gather token rows into bf16 LDS (M_TILE/8 rows per wave)
#pragma unroll
    for (int rr = 0; rr < M_TILE / 8; ++rr) {
        int r = wave * (M_TILE / 8) + rr;
        int t = (r < rows) ? tok_id[e * T + m0 + r] : -1;
        const XT* src = x + (size_t)(t < 0 ? 0 : t) * HIDDEN;
        if constexpr (sizeof(XT) == 4) {   // fp32 source: convert
            const float4* s4 = (const float4*)src;
            for (int c4 = lane; c4 < HIDDEN / 4; c4 += 32) {
                float4 f = (t >= 0) ? s4[c4] : make_float4(0.f, 0.f, 0.f, 0.f);
                v4bf b;
                b[0] = (__bf16)f.x; b[1] = (__bf16)f.y; b[2] = (__bf16)f.z; b[3] = (__bf16)f.w;
                *(v4bf*)(xs + r * XS_STRIDE + c4 * 4) = b;
            }
        } else {                           // bf16 source: straight copy
            const v8bf* s8 = (const v8bf*)src;
            for (int c8 = lane; c8 < HIDDEN / 8; c8 += 32) {
                v8bf b = (t >= 0) ? s8[c8] : v8bf{};
                *(v8bf*)(xs + r * XS_STRIDE + c8 * 8) = b;
            }
        }
    }

    // per-subtile lane base pointers (row + K-phase folded in) so the k-loop
    // addressing stays within the 16-bit DS immediate-offset window per ms
    const __bf16* aBaseX[MSUB];
    const __bf16* aBaseA[MSUB];
#pragma unroll
    for (int ms = 0; ms < MSUB; ++ms) {
        aBaseX[ms] = xs   + (ms * 16 + laneLo) * XS_STRIDE  + laneHi * 16;
        aBaseA[ms] = acts + (ms * 16 + laneLo) * ACT_STRIDE + laneHi * 16;
    }
    __syncthreads();

    // ---- Stage A: gu = x @ w13^T, act = silu(g)*u ----
    const WT* w13e = w13 + (size_t)e * (2 * INTER) * HIDDEN;
#pragma unroll 1
    for (int jt = 0; jt < (INTER / 16) / 8; ++jt) {     // 4 gate/up tile-pairs per wave
        const int jg = wave * 4 + jt;
        const WT* wg = w13e + (size_t)(jg * 16) * HIDDEN;
        const WT* wu = w13e + (size_t)(INTER + jg * 16) * HIDDEN;
        v8f aG[MSUB] = {}, aU[MSUB] = {};
#pragma unroll 2
        for (int kk = 0; kk < HIDDEN / 32; ++kk) {
            v16bf bg = load_b_frag(wg, HIDDEN, kk, laneLo, laneHi);
            v16bf bu = load_b_frag(wu, HIDDEN, kk, laneLo, laneHi);
#pragma unroll
            for (int ms = 0; ms < MSUB; ++ms) {
                v16bf a = load_a_frag_at(aBaseX[ms], kk);
                aG[ms] = __builtin_amdgcn_wmma_f32_16x16x32_bf16(false, a, false, bg, (short)0, aG[ms], false, false);
                aU[ms] = __builtin_amdgcn_wmma_f32_16x16x32_bf16(false, a, false, bu, (short)0, aU[ms], false, false);
            }
        }
        const int col = jg * 16 + laneLo;
#pragma unroll
        for (int ms = 0; ms < MSUB; ++ms) {
#pragma unroll
            for (int v = 0; v < 8; ++v) {
                int row = ms * 16 + v + 8 * laneHi;
                float g = aG[ms][v], u = aU[ms][v];
                float sig = __builtin_amdgcn_rcpf(1.f + __expf(-g));   // fast sigmoid
                acts[row * ACT_STRIDE + col] = (__bf16)(g * sig * u);
            }
        }
    }
    __syncthreads();

    // ---- Stage B: y = act @ w2^T, weighted atomic scatter ----
    const WT* w2e = w2 + (size_t)e * HIDDEN * INTER;
#pragma unroll 1
    for (int nt = 0; nt < (HIDDEN / 16) / 8; ++nt) {    // 12 output tiles per wave
        const int n2 = wave * 12 + nt;
        const WT* wb = w2e + (size_t)(n2 * 16) * INTER;
        v8f c[MSUB] = {};
#pragma unroll 2
        for (int kk = 0; kk < INTER / 32; ++kk) {
            v16bf b = load_b_frag(wb, INTER, kk, laneLo, laneHi);
#pragma unroll
            for (int ms = 0; ms < MSUB; ++ms) {
                v16bf a = load_a_frag_at(aBaseA[ms], kk);
                c[ms] = __builtin_amdgcn_wmma_f32_16x16x32_bf16(false, a, false, b, (short)0, c[ms], false, false);
            }
        }
        const int col = n2 * 16 + laneLo;
#pragma unroll
        for (int ms = 0; ms < MSUB; ++ms) {
#pragma unroll
            for (int v = 0; v < 8; ++v) {
                int row = ms * 16 + v + 8 * laneHi;
                if (row < rows)
                    unsafeAtomicAdd(out + (size_t)lt[row] * HIDDEN + col, lw[row] * c[ms][v]);
            }
        }
    }
}

// ---------------- host launcher ----------------

extern "C" void kernel_launch(void* const* d_in, const int* in_sizes, int n_in,
                              void* d_out, int out_size, void* d_ws, size_t ws_size,
                              hipStream_t stream) {
    const float* x      = (const float*)d_in[0];
    const float* gate_w = (const float*)d_in[1];
    const float* w13    = (const float*)d_in[2];
    const float* w2     = (const float*)d_in[3];
    float* out = (float*)d_out;
    const int T = in_sizes[0] / HIDDEN;   // 8192

    // workspace layout (256B-aligned sections)
    const size_t off_tid  = 256;
    const size_t off_tw   = off_tid + (size_t)NUM_EXPERTS * T * sizeof(int);
    const size_t off_xb   = off_tw  + (size_t)NUM_EXPERTS * T * sizeof(float);
    const size_t n_x      = (size_t)T * HIDDEN;
    const size_t n_w13    = (size_t)NUM_EXPERTS * 2 * INTER * HIDDEN;
    const size_t n_w2     = (size_t)NUM_EXPERTS * HIDDEN * INTER;
    const size_t off_w13b = off_xb   + n_x  * 2;
    const size_t off_w2b  = off_w13b + n_w13 * 2;
    const size_t ws_need_xb  = off_w13b;
    const size_t ws_need_all = off_w2b + n_w2 * 2;

    int*    counts = (int*)d_ws;
    int*    tok_id = (int*)((char*)d_ws + off_tid);
    float*  tok_w  = (float*)((char*)d_ws + off_tw);
    __bf16* xb     = (__bf16*)((char*)d_ws + off_xb);
    __bf16* w13b   = (__bf16*)((char*)d_ws + off_w13b);
    __bf16* w2b    = (__bf16*)((char*)d_ws + off_w2b);

    const size_t n = (size_t)out_size;
    moe_zero_kernel<<<(unsigned)((n + 255) / 256), 256, 0, stream>>>(out, n, counts);
    moe_router_kernel<<<(T + 7) / 8, 256, 0, stream>>>(x, gate_w, counts, tok_id, tok_w, T);

    dim3 grid((T + M_TILE - 1) / M_TILE, NUM_EXPERTS);
    const size_t lds = (size_t)M_TILE * XS_STRIDE * 2 + (size_t)M_TILE * ACT_STRIDE * 2
                     + (size_t)M_TILE * (sizeof(int) + sizeof(float));

    if (ws_size >= ws_need_all) {
        // full bf16 path: one HBM pass to halve all L2-side weight/activation traffic
        moe_cvt_bf16_kernel<<<(unsigned)((n_x   / 8 + 255) / 256), 256, 0, stream>>>(xb,   x,   n_x / 8);
        moe_cvt_bf16_kernel<<<(unsigned)((n_w13 / 8 + 255) / 256), 256, 0, stream>>>(w13b, w13, n_w13 / 8);
        moe_cvt_bf16_kernel<<<(unsigned)((n_w2  / 8 + 255) / 256), 256, 0, stream>>>(w2b,  w2,  n_w2 / 8);
        moe_expert_kernel<__bf16, __bf16><<<grid, 256, lds, stream>>>(xb, w13b, w2b, counts, tok_id, tok_w, out, T);
    } else if (ws_size >= ws_need_xb) {
        moe_cvt_bf16_kernel<<<(unsigned)((n_x / 8 + 255) / 256), 256, 0, stream>>>(xb, x, n_x / 8);
        moe_expert_kernel<__bf16, float><<<grid, 256, lds, stream>>>(xb, w13, w2, counts, tok_id, tok_w, out, T);
    } else {
        moe_expert_kernel<float, float><<<grid, 256, lds, stream>>>(x, w13, w2, counts, tok_id, tok_w, out, T);
    }
}
